// LSTMAllocator_64957085385397
// MI455X (gfx1250) — compile-verified, matrix operand order checked by reading, not run on hardware
//
#include <hip/hip_runtime.h>
#include <hip/hip_bf16.h>

typedef __attribute__((ext_vector_type(16))) _Float16 v16h;
typedef __attribute__((ext_vector_type(8)))  _Float16 v8h;
typedef __attribute__((ext_vector_type(8)))  float    v8f;

#define HDIM 128
#define GH   512        // 4*H gate columns
#define MROWS 16        // batch rows per workgroup
#define LSEQ 256
#define NBLK 128        // 2048 rows / 16

__device__ __forceinline__ float sigf(float v) { return 1.0f / (1.0f + __expf(-v)); }
__device__ __forceinline__ float tanhfast(float v) {
    float e = __expf(2.0f * v);
    return (e - 1.0f) / (e + 1.0f);
}

__global__ void __launch_bounds__(1024)
lstm2_wmma_kernel(const float* __restrict__ x,
                  const float* __restrict__ W_ih0, const float* __restrict__ W_hh0,
                  const float* __restrict__ b_ih0, const float* __restrict__ b_hh0,
                  const float* __restrict__ W_ih1, const float* __restrict__ W_hh1,
                  const float* __restrict__ b_ih1, const float* __restrict__ b_hh1,
                  const float* __restrict__ W1,   const float* __restrict__ b1,
                  const float* __restrict__ W2,   const float* __restrict__ b2,
                  const float* __restrict__ asset_bias,
                  float* __restrict__ scores)
{
    // LDS: h-state (f16 A-fragments, [h1 | h2]), gate accumulators (f32), folded biases
    __shared__ _Float16 hbuf[MROWS][2 * HDIM];   // 8 KB
    __shared__ float    gbuf[MROWS][GH];         // 32 KB
    __shared__ float    biasg0[GH], biasg1[GH], wcol0[GH]; // 6 KB

    const int tid  = threadIdx.x;
    const int lane = tid & 31;
    const int wv   = tid >> 5;            // wave 0..31 -> gate-column strip
    const int mrow = lane & 15;           // M (or N) index within 16x16 tile
    const int koffBase = (lane < 16) ? 0 : 8;
    const int bn0  = blockIdx.x * MROWS;  // first BN row of this workgroup

    // K offset inside a 32-wide K block for half-slot q of a 16-bit A/B fragment
    auto koff = [&](int q) { return (q < 8 ? q : 8 + q) + koffBase; };

    // ---- one-time: build register-resident f16 B fragments (weights) ----
    const int col = wv * 16 + mrow;       // gate column 0..511
    v16h B0[4];                            // W_hh0^T strip, K = 128
    v16h B1[8];                            // [W_ih1 | W_hh1]^T strip, K = 256
    for (int kb = 0; kb < 4; ++kb)
        for (int q = 0; q < 16; ++q)
            B0[kb][q] = (_Float16)W_hh0[col * HDIM + kb * 32 + koff(q)];
    for (int kb = 0; kb < 8; ++kb)
        for (int q = 0; q < 16; ++q) {
            int kk = kb * 32 + koff(q);
            float wval = (kk < HDIM) ? W_ih1[col * HDIM + kk]
                                     : W_hh1[col * HDIM + kk - HDIM];
            B1[kb][q] = (_Float16)wval;
        }

    // ---- init LDS state ----
    for (int i = tid; i < GH; i += 1024) {
        biasg0[i] = b_ih0[i] + b_hh0[i];
        biasg1[i] = b_ih1[i] + b_hh1[i];
        wcol0[i]  = W_ih0[i];             // W_ih0 is [4H x 1]
    }
    for (int i = tid; i < MROWS * 2 * HDIM; i += 1024)
        (&hbuf[0][0])[i] = (_Float16)0.0f;

    float c1[2] = {0.0f, 0.0f};           // cell state layer0 (2 units/thread)
    float c2[2] = {0.0f, 0.0f};           // cell state layer1
    __syncthreads();

    // Load a 16x32 f16 A fragment starting at hbuf column kcol0 (16B-aligned ds_load_b128 x2)
    auto loadA = [&](int kcol0) {
        union { v16h v; v8h h[2]; } u;
        u.h[0] = *(const v8h*)&hbuf[mrow][kcol0 + koffBase];
        u.h[1] = *(const v8h*)&hbuf[mrow][kcol0 + 16 + koffBase];
        return u.v;
    };

    // ---- recurrence over time ----
    // Intra-wave the WMMA chain is serial (accumulator dependence); latency is
    // hidden by the 8 co-resident waves/SIMD of this 32-wave workgroup, which
    // is why the wave stays within the 128-VGPR / 8-wave occupancy budget.
    for (int t = 0; t < LSEQ; ++t) {
        // GEMM phase 0: gates0[:, strip] = h1 @ W_hh0^T   (K = 128)
        {
            v8f acc = {};
#pragma unroll
            for (int kb = 0; kb < 4; ++kb) {
                v16h a = loadA(kb * 32);
                acc = __builtin_amdgcn_wmma_f32_16x16x32_f16(
                    false, a, false, B0[kb], (short)0, acc, false, false);
            }
#pragma unroll
            for (int r = 0; r < 8; ++r)
                gbuf[r + koffBase][col] = acc[r];
        }
        __syncthreads();

        // Update phase 0: nonlinearity, c1/h1
#pragma unroll
        for (int p = 0; p < 2; ++p) {
            int idx = tid + 1024 * p;          // 0..2047 over 16 rows x 128 units
            int row = idx >> 7, u = idx & 127;
            float xt = x[(bn0 + row) * LSEQ + t];
            float gi = gbuf[row][u      ] + xt * wcol0[u      ] + biasg0[u      ];
            float gf = gbuf[row][u + 128] + xt * wcol0[u + 128] + biasg0[u + 128];
            float gg = gbuf[row][u + 256] + xt * wcol0[u + 256] + biasg0[u + 256];
            float go = gbuf[row][u + 384] + xt * wcol0[u + 384] + biasg0[u + 384];
            float c  = sigf(gf) * c1[p] + sigf(gi) * tanhfast(gg);
            c1[p] = c;
            hbuf[row][u] = (_Float16)(sigf(go) * tanhfast(c));
        }
        __syncthreads();

        // GEMM phase 1: gates1[:, strip] = [h1 | h2] @ [W_ih1 | W_hh1]^T   (K = 256)
        {
            v8f acc = {};
#pragma unroll
            for (int kb = 0; kb < 8; ++kb) {
                v16h a = loadA(kb * 32);
                acc = __builtin_amdgcn_wmma_f32_16x16x32_f16(
                    false, a, false, B1[kb], (short)0, acc, false, false);
            }
#pragma unroll
            for (int r = 0; r < 8; ++r)
                gbuf[r + koffBase][col] = acc[r];
        }
        __syncthreads();

        // Update phase 1: c2/h2
#pragma unroll
        for (int p = 0; p < 2; ++p) {
            int idx = tid + 1024 * p;
            int row = idx >> 7, u = idx & 127;
            float gi = gbuf[row][u      ] + biasg1[u      ];
            float gf = gbuf[row][u + 128] + biasg1[u + 128];
            float gg = gbuf[row][u + 256] + biasg1[u + 256];
            float go = gbuf[row][u + 384] + biasg1[u + 384];
            float c  = sigf(gf) * c2[p] + sigf(gi) * tanhfast(gg);
            c2[p] = c;
            hbuf[row][HDIM + u] = (_Float16)(sigf(go) * tanhfast(c));
        }
        __syncthreads();
    }

    // ---- MLP head: relu(h2 @ W1^T + b1) via WMMA (waves 0..7, K=128) ----
    if (wv < 8) {
        const int j = wv * 16 + mrow;      // output column 0..127
        v16h BW[4];
        for (int kb = 0; kb < 4; ++kb)
            for (int q = 0; q < 16; ++q)
                BW[kb][q] = (_Float16)W1[j * HDIM + kb * 32 + koff(q)];
        v8f acc = {};
#pragma unroll
        for (int kb = 0; kb < 4; ++kb) {
            v16h a = loadA(HDIM + kb * 32);   // h2 half of hbuf
            acc = __builtin_amdgcn_wmma_f32_16x16x32_f16(
                false, a, false, BW[kb], (short)0, acc, false, false);
        }
#pragma unroll
        for (int r = 0; r < 8; ++r) {
            float v = acc[r] + b1[j];
            gbuf[r + koffBase][j] = v > 0.0f ? v : 0.0f;
        }
    }
    __syncthreads();

    // ---- score = act @ W2^T + b2 + asset_bias (wave-shuffle reduction) ----
    if (tid < 512) {
        int row = tid >> 5;                 // 0..15
        float part = 0.0f;
#pragma unroll
        for (int q = 0; q < 4; ++q) {
            int j = lane + 32 * q;
            part += gbuf[row][j] * W2[j];
        }
        for (int m = 16; m >= 1; m >>= 1)
            part += __shfl_xor(part, m, 32);
        if (lane == 0) {
            int bn = bn0 + row;
            scores[bn] = part + b2[0] + asset_bias[bn & 63];
        }
    }
}

__global__ void __launch_bounds__(64)
softmax64_kernel(const float* __restrict__ s, float* __restrict__ out)
{
    __shared__ float red[64];
    int b = blockIdx.x, n = threadIdx.x;
    float v = s[b * 64 + n];
    red[n] = v;
    __syncthreads();
    for (int st = 32; st >= 1; st >>= 1) {
        if (n < st) red[n] = fmaxf(red[n], red[n + st]);
        __syncthreads();
    }
    float mx = red[0];
    __syncthreads();
    float e = __expf(v - mx);
    red[n] = e;
    __syncthreads();
    for (int st = 32; st >= 1; st >>= 1) {
        if (n < st) red[n] += red[n + st];
        __syncthreads();
    }
    out[b * 64 + n] = e / red[0];
}

extern "C" void kernel_launch(void* const* d_in, const int* in_sizes, int n_in,
                              void* d_out, int out_size, void* d_ws, size_t ws_size,
                              hipStream_t stream) {
    (void)in_sizes; (void)n_in; (void)out_size; (void)ws_size;
    const float* x      = (const float*)d_in[0];
    const float* W_ih0  = (const float*)d_in[1];
    const float* W_hh0  = (const float*)d_in[2];
    const float* b_ih0  = (const float*)d_in[3];
    const float* b_hh0  = (const float*)d_in[4];
    const float* W_ih1  = (const float*)d_in[5];
    const float* W_hh1  = (const float*)d_in[6];
    const float* b_ih1  = (const float*)d_in[7];
    const float* b_hh1  = (const float*)d_in[8];
    const float* W1     = (const float*)d_in[9];
    const float* b1     = (const float*)d_in[10];
    const float* W2     = (const float*)d_in[11];
    const float* b2     = (const float*)d_in[12];
    const float* abias  = (const float*)d_in[13];
    float* scores = (float*)d_ws;             // 2048 floats of scratch
    float* out    = (float*)d_out;

    lstm2_wmma_kernel<<<dim3(NBLK), dim3(1024), 0, stream>>>(
        x, W_ih0, W_hh0, b_ih0, b_hh0, W_ih1, W_hh1, b_ih1, b_hh1,
        W1, b1, W2, b2, abias, scores);
    softmax64_kernel<<<dim3(32), dim3(64), 0, stream>>>(scores, out);
}
